// Main_GCNN_8589934592116
// MI455X (gfx1250) — compile-verified, matrix operand order checked by reading, not run on hardware
//
#include <hip/hip_runtime.h>

// ---------------------------------------------------------------------------
// MI455X (gfx1250) implementation of the seismic GNN U-net forward pass.
// Memory-bound model (~0.3 TFLOP vs ~2.5 GB traffic -> ~110us HBM floor at
// 23.3 TB/s). All matmul-shaped work runs on v_wmma_f32_16x16x32_f16 (f32 I/O,
// f16 fragments staged in LDS, f32 accumulate), bias/BN/ReLU fused into the
// epilogue, decoder concats are virtual (dual-input conv), and the GEMM B-tile
// uses gfx1250 async global->LDS DMA when available.
// ---------------------------------------------------------------------------

typedef __attribute__((ext_vector_type(16))) _Float16 v16h;
typedef __attribute__((ext_vector_type(8)))  float    v8f;
typedef int v4i_ __attribute__((vector_size(16)));

#if __has_builtin(__builtin_amdgcn_global_load_async_to_lds_b128)
#define HAVE_ASYNC_LDS 1
#else
#define HAVE_ASYNC_LDS 0
#endif

static constexpr int Nn   = 1024;   // G * NPG
static constexpr int Lw   = 3072;
static constexpr int Gg   = 32;
static constexpr int NPGc = 32;

static inline int ceilDiv(int a, int b) { return (a + b - 1) / b; }

// K-index mapping inside a 16-bit 16x32 WMMA fragment (ISA 7.12.2):
// lanes 0-15 hold K=[0..7]+[16..23] in elements 0..15, lanes 16-31 hold +8.
__device__ __forceinline__ int kmap(int e, int hi) {
  return (e < 8 ? e : e + 8) + hi * 8;
}

// ---------------------------------------------------------------------------
// Fused Conv1d / ConvTranspose1d (k=7, pad=3) + bias + BN + ReLU as implicit
// GEMM. One block = one node x 16 out-channels x TILE time positions.
// LDS layout: lut[Kpad] (int: im2row row offset per K), Wperm (weights in
// fragment order -> v16h vector loads), Xs[(Cin+1)][TILE+6] (haloed patch,
// last row all-zero so padded K needs no branch).
// ---------------------------------------------------------------------------
__global__ void k_conv_wmma(const float* __restrict__ X, long xStride, int C1,
                            const float* __restrict__ X2, long x2Stride,
                            const float* __restrict__ W, const float* __restrict__ bias,
                            const float* __restrict__ bnG, const float* __restrict__ bnB,
                            const float* __restrict__ bnM, const float* __restrict__ bnV,
                            float* __restrict__ Y, long yStride, int chOff,
                            int Cin, int Lin, int Cout, int up, int relu, int wT,
                            int TILE)
{
  extern __shared__ __align__(32) char smemRaw[];
  const int K    = Cin * 7;
  const int Kpad = (K + 31) & ~31;
  const int PW   = TILE + 6;
  int*      lut   = (int*)smemRaw;                       // [Kpad]
  _Float16* Wperm = (_Float16*)(smemRaw + Kpad * 4);     // [Kpad/32][32][16]
  _Float16* Xs    = Wperm + 16 * Kpad;                   // [(Cin+1)][PW]
  const int node = blockIdx.z;
  const int coT  = blockIdx.y;
  const int t0   = blockIdx.x * TILE;
  const int Lout = Lin * up;

  // im2row lookup: K -> row offset into Xs (zero pad row for K >= Cin*7).
  for (int kk = threadIdx.x; kk < Kpad; kk += blockDim.x)
    lut[kk] = (kk < K) ? (kk / 7) * PW + (kk % 7) : Cin * PW;

  // Stage weights in WMMA A-fragment order (f32 -> f16).
  for (int idx = threadIdx.x; idx < 16 * Kpad; idx += blockDim.x) {
    int stepi = idx >> 9, r = idx & 511;
    int l = r >> 4, e = r & 15;
    int m = l & 15, hi = l >> 4;
    int kk = stepi * 32 + kmap(e, hi);
    int co = coT * 16 + m;
    float v = 0.f;
    if (kk < K && co < Cout) {
      int ci = kk / 7, dk = kk - ci * 7;
      v = wT ? W[((long)ci * Cout + co) * 7 + (6 - dk)]
             : W[((long)co * Cin + ci) * 7 + dk];
    }
    Wperm[idx] = (_Float16)v;
  }
  // Stage input patch (zero-stuffed for conv-transpose; last row = zeros).
  for (int idx = threadIdx.x; idx < (Cin + 1) * PW; idx += blockDim.x) {
    int ci = idx / PW, tt = idx - ci * PW;
    int tg = t0 + tt - 3;
    float v = 0.f;
    if (ci < Cin && tg >= 0 && tg < Lout && (up == 1 || (tg % up) == 0)) {
      const float* src = (ci < C1)
          ? (X  + (long)node * xStride  + (long)ci        * Lin)
          : (X2 + (long)node * x2Stride + (long)(ci - C1) * Lin);
      v = src[tg / up];
    }
    Xs[idx] = (_Float16)v;
  }
  __syncthreads();

  const int wave = threadIdx.x >> 5;
  const int lane = threadIdx.x & 31;
  const int n = lane & 15, hi = lane >> 4;
  const int col = wave * 16 + n;
  v8f acc = {};
  auto stepf = [&](int k0) {
    v16h a = *(const v16h*)&Wperm[((k0 >> 5) * 32 + lane) * 16];
    v16h b;
#pragma unroll
    for (int e = 0; e < 16; ++e)
      b[e] = Xs[lut[k0 + kmap(e, hi)] + col];
    acc = __builtin_amdgcn_wmma_f32_16x16x32_f16(false, a, false, b,
                                                 (short)0, acc, false, false);
  };
  int k0 = 0;
  for (; k0 + 64 <= Kpad; k0 += 64) { stepf(k0); stepf(k0 + 32); }
  for (; k0 < Kpad; k0 += 32) stepf(k0);

  // Epilogue: bias + BN + ReLU, per ISA C/D layout (row = r + hi*8).
#pragma unroll
  for (int r = 0; r < 8; ++r) {
    int co = coT * 16 + r + hi * 8;
    int t  = t0 + wave * 16 + n;
    if (co < Cout && t < Lout) {
      float v = acc[r] + (bias ? bias[co] : 0.f);
      if (bnG) {
        float s = bnG[co] * rsqrtf(bnV[co] + 1e-5f);
        v = (v - bnM[co]) * s + bnB[co];
      }
      if (relu) v = fmaxf(v, 0.f);
      Y[(long)node * yStride + (long)(chOff + co) * Lout + t] = v;
    }
  }
}

// ---------------------------------------------------------------------------
// Dense GEMM  C[M,Nc] = A[M,K] * W[K,Nc] + b  (f16 WMMA, f32 accumulate).
// 256 threads (8 waves), tile 16(M) x 128(N), BK=64 (2 WMMA per barrier).
// Full interior B-tiles are DMA'd raw (f32) into LDS with
// global_load_async_to_lds_b128 + s_wait_asynccnt, then converted in LDS.
// ---------------------------------------------------------------------------
__global__ void k_gemm_wmma(const float* __restrict__ A, const float* __restrict__ W,
                            const float* __restrict__ bias, float* __restrict__ C,
                            int M, int K, int Nc, long ldc, int colOff, int act)
{
  __shared__ __align__(32) _Float16 Aperm[2 * 32 * 16];  // fragment-order A
  __shared__ _Float16 Bs[64 * 128];                      // f16 B tile [k][c]
  __shared__ float    Bst[64 * 128];                     // raw f32 async landing
  const int mT = blockIdx.y, nT = blockIdx.x;
  const int wave = threadIdx.x >> 5, lane = threadIdx.x & 31;
  const int n = lane & 15, hi = lane >> 4;
  v8f acc = {};
  for (int k0 = 0; k0 < K; k0 += 64) {
    __syncthreads();
    // A tile -> fragment order (two 32-wide K substeps).
    for (int idx = threadIdx.x; idx < 2 * 32 * 16; idx += blockDim.x) {
      int sub = idx >> 9, r = idx & 511;
      int l = r >> 4, e = r & 15;
      int gm = mT * 16 + (l & 15);
      int kk = k0 + sub * 32 + kmap(e, l >> 4);
      Aperm[idx] = (_Float16)((gm < M && kk < K) ? A[(long)gm * K + kk] : 0.f);
    }
#if HAVE_ASYNC_LDS
    if (k0 + 64 <= K && (nT + 1) * 128 <= Nc) {
      // Each wave DMAs 8 rows of 128 f32 (one b128 per lane per row).
      const float* gbase = W + (long)(k0 + wave * 8) * Nc + (long)nT * 128 + lane * 4;
      for (int r = 0; r < 8; ++r) {
        void* gsrc = const_cast<void*>((const void*)(gbase + (long)r * Nc));
        void* ldst = (void*)&Bst[(wave * 8 + r) * 128 + lane * 4];
        __builtin_amdgcn_global_load_async_to_lds_b128(
            (__attribute__((address_space(1))) v4i_*)gsrc,
            (__attribute__((address_space(3))) v4i_*)ldst,
            0, 0);
      }
#if __has_builtin(__builtin_amdgcn_s_wait_asynccnt)
      __builtin_amdgcn_s_wait_asynccnt(0);
#endif
      __syncthreads();
      for (int idx = threadIdx.x; idx < 64 * 128; idx += blockDim.x)
        Bs[idx] = (_Float16)Bst[idx];
    } else
#endif
    {
      for (int idx = threadIdx.x; idx < 64 * 128; idx += blockDim.x) {
        int kk = idx >> 7, c = idx & 127;
        int gk = k0 + kk, gc = nT * 128 + c;
        Bs[idx] = (_Float16)((gk < K && gc < Nc) ? W[(long)gk * Nc + gc] : 0.f);
      }
    }
    __syncthreads();
#pragma unroll
    for (int sub = 0; sub < 2; ++sub) {
      v16h a = *(const v16h*)&Aperm[(sub * 32 + lane) * 16];
      v16h b;
#pragma unroll
      for (int e = 0; e < 16; ++e)
        b[e] = Bs[(sub * 32 + kmap(e, hi)) * 128 + wave * 16 + n];
      acc = __builtin_amdgcn_wmma_f32_16x16x32_f16(false, a, false, b,
                                                   (short)0, acc, false, false);
    }
  }
#pragma unroll
  for (int r = 0; r < 8; ++r) {
    int gm = mT * 16 + r + hi * 8;
    int gc = nT * 128 + wave * 16 + n;
    if (gm < M && gc < Nc) {
      float v = acc[r] + (bias ? bias[gc] : 0.f);
      if (act == 1) v = fmaxf(v, 0.f);
      C[(long)gm * ldc + colOff + gc] = v;
    }
  }
}

// ---------------------------------------------------------------------------
// TransformerConv aggregation. Node i's 16 incoming edges are contiguous at
// src[i*16..], so segment-softmax == per-node softmax. One wave32 per node;
// Out already holds x@Ws+bs and gets += softmax-weighted V aggregation.
// ---------------------------------------------------------------------------
__global__ void k_attn16(const float* __restrict__ Q, const float* __restrict__ Kf,
                         const float* __restrict__ V, const int* __restrict__ src,
                         float* __restrict__ Out, int nNodes, int d)
{
  int node = blockIdx.x * (blockDim.x >> 5) + (threadIdx.x >> 5);
  int lane = threadIdx.x & 31;
  if (node >= nNodes) return;
  const float scale = rsqrtf((float)d);
  float logit[16];
  float mx = -1e30f;
  for (int e = 0; e < 16; ++e) {
    int s = src[node * 16 + e];
    float p = 0.f;
    for (int c = lane; c < d; c += 32)
      p += Q[(long)node * d + c] * Kf[(long)s * d + c];
    for (int off = 16; off; off >>= 1) p += __shfl_xor(p, off, 32);
    p *= scale;
    logit[e] = p;
    mx = fmaxf(mx, p);
  }
  float w[16], z = 0.f;
  for (int e = 0; e < 16; ++e) { w[e] = __expf(logit[e] - mx); z += w[e]; }
  float inv = 1.f / fmaxf(z, 1e-16f);
  for (int c = lane; c < d; c += 32) {
    float a = 0.f;
    for (int e = 0; e < 16; ++e)
      a += w[e] * V[(long)src[node * 16 + e] * d + c];
    Out[(long)node * d + c] += a * inv;
  }
}

// ---- glue kernels ----------------------------------------------------------
__global__ void k_maxpool5(const float* __restrict__ X, float* __restrict__ Y,
                           int C, int Lin, int stride)
{
  int Lout = Lin / stride;
  long i = (long)blockIdx.x * blockDim.x + threadIdx.x;
  if (i >= (long)C * Lout) return;
  int c = (int)(i / Lout), t = (int)(i - (long)c * Lout);
  const float* xp = X + ((long)blockIdx.z * C + c) * Lin;
  float m = -1e30f;
  int base = t * stride - 2;
  for (int j = 0; j < 5; ++j) {
    int p = base + j;
    if (p >= 0 && p < Lin) m = fmaxf(m, xp[p]);
  }
  Y[((long)blockIdx.z * C + c) * Lout + t] = m;
}

__global__ void k_act(float* __restrict__ X, long n, int act)  // 2=sigmoid,3=tanh
{
  long i = (long)blockIdx.x * blockDim.x + threadIdx.x;
  if (i >= n) return;
  float v = X[i];
  X[i] = (act == 2) ? 1.f / (1.f + __expf(-v)) : tanhf(v);
}

__global__ void k_graph_mean_sig(const float* __restrict__ xd, float* __restrict__ out,
                                 int nG, int npg)
{
  int g = blockIdx.x * blockDim.x + threadIdx.x;
  if (g >= nG) return;
  float s = 0.f;
  for (int i = 0; i < npg; ++i) s += xd[g * npg + i];
  s /= (float)npg;
  out[g] = 1.f / (1.f + __expf(-s));
}

__global__ void k_stack_t(const float* __restrict__ off, const float* __restrict__ dep,
                          float* __restrict__ t2, int nNodes, int npg)
{
  int i = blockIdx.x * blockDim.x + threadIdx.x;
  if (i >= nNodes) return;
  t2[i * 2 + 0] = off[i];
  t2[i * 2 + 1] = dep[i / npg];
}

__global__ void k_shift(const float* __restrict__ d, int* __restrict__ s, int n, int Ln)
{
  int i = blockIdx.x * blockDim.x + threadIdx.x;
  if (i < n) s[i] = (int)(d[i] * (float)Ln);   // C truncation == astype(int32)
}

__global__ void k_roll(const float* __restrict__ X, long xRow,
                       const int* __restrict__ sh, int sgn,
                       float* __restrict__ Y, long yRow, int Ln)
{
  int i = blockIdx.y;
  int t = blockIdx.x * blockDim.x + threadIdx.x;
  if (t >= Ln) return;
  long idx = ((long)t - (long)sgn * sh[i]) % Ln;
  if (idx < 0) idx += Ln;
  Y[(long)i * yRow + t] = X[(long)i * xRow + idx];
}

// ---------------------------------------------------------------------------
// Host orchestration.
// ---------------------------------------------------------------------------
struct CD  { const float *b1,*b2,*bn1g,*bn1b,*bn1m,*bn1v,*bn2g,*bn2b,*bn2m,*bn2v,*w1,*w2; };
struct CT  { const float *b,*bng,*bnb,*bnm,*bnv,*w; };
struct TCp { const float *Wk,*Wq,*Ws,*Wv,*bk,*bq,*bs,*bv; };
struct MLPp{ const float *b1,*b2,*w1,*w2; };
struct O1p { const float *b,*bng,*bnb,*bnm,*bnv,*w; };
struct O2p { const float *b,*w; };

extern "C" void kernel_launch(void* const* d_in, const int* in_sizes, int n_in,
                              void* d_out, int out_size, void* d_ws, size_t ws_size,
                              hipStream_t stream)
{
  (void)in_sizes; (void)n_in; (void)out_size; (void)ws_size;
  const float* x    = (const float*)d_in[0];   // [N,3,L]
  const float* sloc = (const float*)d_in[1];   // [N,3]
  const int*   esrc = (const int*)  d_in[3];   // edge_index row 0 (src), [E]

  // ---- params: pytree-flatten order (sorted keys, tuples in order) ----
  int pi = 4;
  auto nx    = [&]() { return (const float*)d_in[pi++]; };
  auto ldCD  = [&]() { CD c; c.b1=nx(); c.b2=nx(); c.bn1g=nx(); c.bn1b=nx(); c.bn1m=nx(); c.bn1v=nx();
                       c.bn2g=nx(); c.bn2b=nx(); c.bn2m=nx(); c.bn2v=nx(); c.w1=nx(); c.w2=nx(); return c; };
  auto ldCT  = [&]() { CT c; c.b=nx(); c.bng=nx(); c.bnb=nx(); c.bnm=nx(); c.bnv=nx(); c.w=nx(); return c; };
  auto ldTC  = [&]() { TCp c; c.Wk=nx(); c.Wq=nx(); c.Ws=nx(); c.Wv=nx(); c.bk=nx(); c.bq=nx(); c.bs=nx(); c.bv=nx(); return c; };
  auto ldMLP = [&]() { MLPp c; c.b1=nx(); c.b2=nx(); c.w1=nx(); c.w2=nx(); return c; };
  auto ldO1  = [&]() { O1p c; c.b=nx(); c.bng=nx(); c.bnb=nx(); c.bnm=nx(); c.bnv=nx(); c.w=nx(); return c; };
  auto ldO2  = [&]() { O2p c; c.b=nx(); c.w=nx(); return c; };

  CD  ah_gd1=ldCD(), ah_gd2=ldCD();
  MLPp ah_smlp=ldMLP();
  TCp ah_tc1=ldTC(), ah_tc2=ldTC(), ah_tc3=ldTC(), ah_tcd=ldTC();
  MLPp ah_tmlp=ldMLP();
  CD  gc1=ldCD(), gc2=ldCD(), gc3=ldCD(), gc4=ldCD(), gc5cd=ldCD();
  TCp gc5p=ldTC(), gc5s=ldTC();
  CD  gd1=ldCD(), gd2=ldCD(), gd3=ldCD(), gd4=ldCD();
  CT  gu1=ldCT(), gu2=ldCT(), gu3=ldCT(), gu4=ldCT();
  O1p o1p=ldO1(), o1s=ldO1();
  O2p o2p=ldO2(), o2s=ldO2();
  TCp th_p1=ldTC(), th_p2=ldTC(), th_s1=ldTC(), th_s2=ldTC();

  // ---- workspace bump allocator ----
  char* wsBase = (char*)d_ws;
  size_t wsOff = 0;
  auto alloc = [&](size_t elems) {
    float* p = (float*)(wsBase + wsOff);
    wsOff = (wsOff + elems * sizeof(float) + 255) & ~(size_t)255;
    return p;
  };

  float* tA    = alloc((size_t)Nn * 16 * Lw);   // generic cd mid-buffer (max size)
  float* x3072 = alloc((size_t)Nn * 16 * Lw);
  float* p1    = alloc((size_t)Nn * 16 * 768);
  float* x768  = alloc((size_t)Nn * 32 * 768);
  float* p2    = alloc((size_t)Nn * 32 * 192);
  float* x192  = alloc((size_t)Nn * 64 * 192);
  float* p3    = alloc((size_t)Nn * 64 * 48);
  float* x48   = alloc((size_t)Nn * 128 * 48);
  float* out0  = alloc((size_t)Nn * 128 * 24);
  float* h32   = alloc((size_t)Nn * 32 * 24);
  float* hp12  = alloc((size_t)Nn * 32 * 12);
  float* hcat  = alloc((size_t)Nn * 192);
  float* hm48  = alloc((size_t)Nn * 48);
  float* hm96  = alloc((size_t)Nn * 96);
  float* h96   = alloc((size_t)Nn * 96);
  float* h2    = alloc((size_t)Nn * 12);
  float* xd    = alloc((size_t)Nn);
  float* t2    = alloc((size_t)Nn * 2);
  float* tmp4  = alloc((size_t)Nn * 4);
  float* qb    = alloc((size_t)Nn * Lw);
  float* kb    = alloc((size_t)Nn * Lw);
  float* vb    = alloc((size_t)Nn * Lw);
  float* dc1   = alloc((size_t)Nn * 128 * 24);
  float* d1    = alloc((size_t)Nn * 128 * 48);
  float* b2a   = alloc((size_t)Nn * 128 * 48);
  float* d2    = alloc((size_t)Nn * 64 * 192);
  float* b3    = alloc((size_t)Nn * 64 * 192);
  float* d3    = alloc((size_t)Nn * 32 * 768);
  float* b4    = alloc((size_t)Nn * 32 * 768);
  float* d4    = alloc((size_t)Nn * 16 * Lw);
  float* val   = alloc((size_t)Nn * 2 * Lw);
  float* rolled= alloc((size_t)Nn * Lw);
  float* gout  = alloc((size_t)Nn * Lw);
  float* galign= alloc((size_t)Nn * Lw);
  float* h1    = alloc((size_t)Nn * Lw);
  int*   spI   = (int*)alloc(Nn);
  int*   ssI   = (int*)alloc(Nn);

  // ---- outputs (concatenated in reference return order) ----
  float* outp = (float*)d_out;
  float* out_p   = outp;
  float* out_s   = outp + (long)Nn * Lw;
  float* auxoff  = outp + 2L * Nn * Lw;
  float* auxdep  = auxoff + Nn;
  float* dtp     = auxdep + Gg;
  float* dts     = dtp + Nn;
  float* xtemp   = dts + Nn;        // [N,48]

  // ---- launchers ----
  auto conv = [&](const float* X0, long xs, int C1, const float* X2, long x2s,
                  const float* W, const float* b,
                  const float* g, const float* be, const float* m, const float* v,
                  float* Y, long ys, int chOff,
                  int Cin, int Lin, int Cout, int up, int relu, int wT) {
    int K = Cin * 7, Kpad = (K + 31) & ~31;
    int TILE = 128;
    auto bytes = [&](int T) {
      return (size_t)Kpad * 4 +
             (size_t)(16 * Kpad + (Cin + 1) * (T + 6)) * sizeof(_Float16);
    };
    while (TILE > 16 && bytes(TILE) > 100 * 1024) TILE >>= 1;
    int Lout = Lin * up;
    dim3 grid(ceilDiv(Lout, TILE), ceilDiv(Cout, 16), Nn);
    dim3 blk((TILE / 16) * 32);
    k_conv_wmma<<<grid, blk, bytes(TILE), stream>>>(X0, xs, C1, X2, x2s, W, b,
        g, be, m, v, Y, ys, chOff, Cin, Lin, Cout, up, relu, wT, TILE);
  };
  auto cdrun = [&](const float* X0, long xs, int C1, const float* X2, long x2s,
                   const CD& p, int Cin, int Co, int Lin, float* Y, long ys, int chOff) {
    conv(X0, xs, C1, X2, x2s, p.w1, p.b1, p.bn1g, p.bn1b, p.bn1m, p.bn1v,
         tA, (long)Co * Lin, 0, Cin, Lin, Co, 1, 1, 0);
    conv(tA, (long)Co * Lin, Co, nullptr, 0, p.w2, p.b2, p.bn2g, p.bn2b, p.bn2m, p.bn2v,
         Y, ys, chOff, Co, Lin, Co, 1, 1, 0);
  };
  auto pool = [&](const float* X0, int C, int Lin, int s, float* Y) {
    int Lout = Lin / s;
    dim3 grid(ceilDiv(C * Lout, 256), 1, Nn);
    k_maxpool5<<<grid, 256, 0, stream>>>(X0, Y, C, Lin, s);
  };
  auto gemm = [&](const float* A, const float* W, const float* b, float* C,
                  int M, int K, int Ncol, long ldc, int colOff, int act) {
    dim3 grid(ceilDiv(Ncol, 128), ceilDiv(M, 16));
    k_gemm_wmma<<<grid, 256, 0, stream>>>(A, W, b, C, M, K, Ncol, ldc, colOff, act);
  };
  auto tf = [&](const float* X0, int din, int dout, const TCp& p, float* out) {
    gemm(X0, p.Wq, p.bq, qb,  Nn, din, dout, dout, 0, 0);
    gemm(X0, p.Wk, p.bk, kb,  Nn, din, dout, dout, 0, 0);
    gemm(X0, p.Wv, p.bv, vb,  Nn, din, dout, dout, 0, 0);
    gemm(X0, p.Ws, p.bs, out, Nn, din, dout, dout, 0, 0);   // root/skip term
    k_attn16<<<ceilDiv(Nn, 8), 256, 0, stream>>>(qb, kb, vb, esrc, out, Nn, dout);
  };

  // ================= encoder =================
  cdrun(x,    3L * Lw,   3,  nullptr, 0, gd1, 3,   16, Lw,  x3072, 16L * Lw,  0);
  pool(x3072, 16, Lw, 4, p1);
  cdrun(p1,   16L * 768, 16, nullptr, 0, gd2, 16,  32, 768, x768,  32L * 768, 0);
  pool(x768,  32, 768, 4, p2);
  cdrun(p2,   32L * 192, 32, nullptr, 0, gd3, 32,  64, 192, x192,  64L * 192, 0);
  pool(x192,  64, 192, 4, p3);
  cdrun(p3,   64L * 48,  64, nullptr, 0, gd4, 64, 128, 48,  x48,   128L * 48, 0);
  pool(x48,  128, 48, 2, out0);

  // ================= location head =================
  cdrun(out0, 128L * 24, 128, nullptr, 0, ah_gd1, 128, 32, 24, h32, 32L * 24, 0);
  pool(h32, 32, 24, 2, hp12);
  cdrun(hp12, 32L * 12, 32, nullptr, 0, ah_gd2, 32, 8, 12, hcat, 192, 0); // -> hcat[:,0:96]
  gemm(sloc, ah_smlp.w1, ah_smlp.b1, hm48, Nn, 3,  48, 48,  0,  1);
  gemm(hm48, ah_smlp.w2, ah_smlp.b2, hcat, Nn, 48, 96, 192, 96, 0);        // -> hcat[:,96:192]
  gemm(hcat, ah_tmlp.w1, ah_tmlp.b1, hm96, Nn, 192, 96, 96, 0, 1);
  gemm(hm96, ah_tmlp.w2, ah_tmlp.b2, h96,  Nn, 96,  96, 96, 0, 0);
  tf(h96,   96, 48, ah_tc1, xtemp);                        // x_temp (output)
  tf(xtemp, 48, 12, ah_tc2, h2);
  tf(h2,    12, 1,  ah_tc3, auxoff);
  k_act<<<ceilDiv(Nn, 256), 256, 0, stream>>>(auxoff, Nn, 2);   // sigmoid
  tf(h2,    12, 1,  ah_tcd, xd);
  k_graph_mean_sig<<<1, Gg, 0, stream>>>(xd, auxdep, Gg, NPGc);

  // ================= time head =================
  k_stack_t<<<ceilDiv(Nn, 256), 256, 0, stream>>>(auxoff, auxdep, t2, Nn, NPGc);
  tf(t2,   2, 4, th_p1, tmp4);
  tf(tmp4, 4, 1, th_p2, dtp);
  k_act<<<ceilDiv(Nn, 256), 256, 0, stream>>>(dtp, Nn, 3);      // tanh
  tf(t2,   2, 4, th_s1, tmp4);
  tf(tmp4, 4, 1, th_s2, dts);
  k_act<<<ceilDiv(Nn, 256), 256, 0, stream>>>(dts, Nn, 3);

  // ================= decoder (virtual skip-concat) =================
  cdrun(out0, 128L * 24, 128, nullptr, 0, gc1, 128, 128, 24, dc1, 128L * 24, 0);
  conv(dc1, 128L * 24, 128, nullptr, 0, gu1.w, gu1.b, gu1.bng, gu1.bnb, gu1.bnm, gu1.bnv,
       d1, 128L * 48, 0, 128, 24, 128, 2, 1, 1);
  cdrun(x48,  128L * 48,  128, d1, 128L * 48,  gc2, 256, 128, 48,  b2a, 128L * 48,  0);
  conv(b2a, 128L * 48, 128, nullptr, 0, gu2.w, gu2.b, gu2.bng, gu2.bnb, gu2.bnm, gu2.bnv,
       d2, 64L * 192, 0, 128, 48, 64, 4, 1, 1);
  cdrun(x192, 64L * 192,  64,  d2, 64L * 192,  gc3, 128, 64,  192, b3,  64L * 192,  0);
  conv(b3, 64L * 192, 64, nullptr, 0, gu3.w, gu3.b, gu3.bng, gu3.bnb, gu3.bnm, gu3.bnv,
       d3, 32L * 768, 0, 64, 192, 32, 4, 1, 1);
  cdrun(x768, 32L * 768,  32,  d3, 32L * 768,  gc4, 64,  32,  768, b4,  32L * 768,  0);
  conv(b4, 32L * 768, 32, nullptr, 0, gu4.w, gu4.b, gu4.bng, gu4.bnb, gu4.bnm, gu4.bnv,
       d4, 16L * Lw, 0, 32, 768, 16, 4, 1, 1);
  cdrun(x3072, 16L * Lw, 16, d4, 16L * Lw, gc5cd, 32, 2, Lw, val, 2L * Lw, 0);

  // ================= GNN_layer_Trans: roll -> TransformerConv -> unroll =====
  k_shift<<<ceilDiv(Nn, 256), 256, 0, stream>>>(dtp, spI, Nn, Lw);
  k_shift<<<ceilDiv(Nn, 256), 256, 0, stream>>>(dts, ssI, Nn, Lw);
  dim3 rg(ceilDiv(Lw, 256), Nn);

  // P branch
  k_roll<<<rg, 256, 0, stream>>>(val + 0 * Lw, 2L * Lw, spI, +1, rolled, Lw, Lw);
  tf(rolled, Lw, Lw, gc5p, gout);                 // 4x [1024,3072]x[3072,3072] GEMMs
  k_roll<<<rg, 256, 0, stream>>>(gout, Lw, spI, -1, galign, Lw, Lw);
  conv(galign, (long)Lw, 1, nullptr, 0, o1p.w, o1p.b, o1p.bng, o1p.bnb, o1p.bnm, o1p.bnv,
       h1, (long)Lw, 0, 1, Lw, 1, 1, 0, 0);
  conv(h1, (long)Lw, 1, nullptr, 0, o2p.w, o2p.b, nullptr, nullptr, nullptr, nullptr,
       out_p, (long)Lw, 0, 1, Lw, 1, 1, 0, 0);

  // S branch
  k_roll<<<rg, 256, 0, stream>>>(val + 1 * Lw, 2L * Lw, ssI, +1, rolled, Lw, Lw);
  tf(rolled, Lw, Lw, gc5s, gout);
  k_roll<<<rg, 256, 0, stream>>>(gout, Lw, ssI, -1, galign, Lw, Lw);
  conv(galign, (long)Lw, 1, nullptr, 0, o1s.w, o1s.b, o1s.bng, o1s.bnb, o1s.bnm, o1s.bnv,
       h1, (long)Lw, 0, 1, Lw, 1, 1, 0, 0);
  conv(h1, (long)Lw, 1, nullptr, 0, o2s.w, o2s.b, nullptr, nullptr, nullptr, nullptr,
       out_s, (long)Lw, 0, 1, Lw, 1, 1, 0, 0);
}
